// DSC_28535762715377
// MI455X (gfx1250) — matrix operand order repaired and checked by reading, not compiled, not run on hardware
//
#include <hip/hip_runtime.h>

// MI455X / gfx1250. Roofline: term4 streams 340MB of M once at ~1 FLOP/byte
// -> pure HBM-bandwidth bound (~14.6us at 23.3 TB/s). Main kernel is an f32
// streaming reduction with coalesced b64 loads (8 tiles = 2KB per wave-iter,
// scalar index math via readfirstlane); the small dense contractions
// (terms 2/3) use v_wmma_f32_16x16x32_f16 with f32 accumulation and fully
// scalar (EXEC-preserving) control flow.

typedef __attribute__((ext_vector_type(16))) _Float16 v16h;
typedef __attribute__((ext_vector_type(8)))  float    v8f;

constexpr int kH  = 24;
constexpr int kMd = 48;
constexpr int kR  = kH * kMd;                   // 1152
constexpr long long kNT = (long long)kR * kR;   // 1,327,104 tiles of 64 floats
constexpr int kNGRP8 = (int)(kNT / 8);          // 165,888 groups of 8 tiles

// ---------------------------------------------------------------------------
// Setup: a[i*48+j] = lam^0.25[i]*phi[j,i];  b[l*48+k] = sig^0.25[l]*phit[k,l]
// Also writes term1 (M_tilde[0,0] @ y_rev[0]) into out (initializes it).
// ---------------------------------------------------------------------------
__global__ void setup_kernel(const float* __restrict__ y,
                             const float* __restrict__ Mt,
                             const float* __restrict__ sigma,
                             const float* __restrict__ lambda_e,
                             const float* __restrict__ phi,
                             const float* __restrict__ phit,
                             float* __restrict__ wa,
                             float* __restrict__ wb,
                             float* __restrict__ out)
{
    int tid = blockIdx.x * blockDim.x + threadIdx.x;
    if (tid < kR) {
        int i = tid / kMd;
        int j = tid - i * kMd;
        wa[tid] = powf(lambda_e[i], 0.25f) * phi[j * kH + i];   // lam4[i]*phi[j,i]
        wb[tid] = powf(sigma[i],    0.25f) * phit[j * kH + i];  // sig4[l]*phit[k,l]
    }
    if (tid < 8) {  // term1: u[c] = sum_p M_tilde[0,0,c,p] * y_rev[0,p]
        float s = 0.f;
        #pragma unroll
        for (int p = 0; p < 8; ++p) s += Mt[tid * 8 + p] * y[p];
        out[tid] = s;
    }
}

// ---------------------------------------------------------------------------
// Term 4: u[c] += sum_{t,p} a[r1]*b[r2] * M[t*64 + c*8 + p] * y[(j+k)*8 + p]
// t = r1*1152 + r2, r1 = i*48+j, r2 = l*48+k. Lane L owns (c=L>>2, p=2*(L&3)).
// 8 consecutive tiles per wave-iteration -> 2KB contiguous global_load_b64.
// Group index is wave-uniform (readfirstlane) -> all /1152, %48 math is SALU
// and loop branches are scalar. Groups of 8 never cross r1 or k%48 boundaries
// (1152 and 48 are multiples of 8).
// ---------------------------------------------------------------------------
__global__ void __launch_bounds__(256)
term4_kernel(const float* __restrict__ M,
             const float* __restrict__ y,
             const float* __restrict__ wa,
             const float* __restrict__ wb,
             float* __restrict__ out)
{
    __shared__ float sY[768];
    __shared__ float sA[kR];
    __shared__ float sB[kR];
    __shared__ float red[8];

    for (int x = threadIdx.x; x < 768; x += blockDim.x) sY[x] = y[x];
    for (int x = threadIdx.x; x < kR;  x += blockDim.x) { sA[x] = wa[x]; sB[x] = wb[x]; }
    if (threadIdx.x < 8) red[threadIdx.x] = 0.f;
    __syncthreads();

    const int gid    = blockIdx.x * blockDim.x + threadIdx.x;
    const int wave   = __builtin_amdgcn_readfirstlane(gid >> 5);     // SGPR
    const int nwaves = (gridDim.x * blockDim.x) >> 5;                // SGPR
    const int lane   = threadIdx.x & 31;
    const int c      = lane >> 2;
    const int p0     = (lane & 3) << 1;

    float acc = 0.f;
    for (int g = wave; g < kNGRP8; g += nwaves) {       // scalar loop
        const int tb = g << 3;                          // base tile (mult. of 8)
        const int r1 = tb / kR;                         // SALU
        const int r2 = tb - r1 * kR;                    // SALU
        const int j  = r1 % kMd;                        // SALU
        const int kb = r2 % kMd;                        // SALU, kb <= 40
        const float a1 = sA[r1];
        const float4 b0 = *reinterpret_cast<const float4*>(&sB[r2]);      // r2%8==0
        const float4 b1 = *reinterpret_cast<const float4*>(&sB[r2 + 4]);
        const float bw[8] = {b0.x, b0.y, b0.z, b0.w, b1.x, b1.y, b1.z, b1.w};
        const float2* mp = reinterpret_cast<const float2*>(M) + (size_t)tb * 32 + lane;
        const float2* yp = reinterpret_cast<const float2*>(&sY[(j + kb) * 8 + p0]);

        // Prefetch next group's 2KB (scalar guard keeps EXEC untouched).
        if (g + nwaves < kNGRP8) {
            const char* pf = reinterpret_cast<const char*>(M)
                           + (((size_t)(g + nwaves)) << 11) + (size_t)lane * 64;
            __builtin_prefetch(pf, 0, 0);               // global_prefetch_b8
        }

        #pragma unroll
        for (int u = 0; u < 8; ++u) {
            const float2 mv = mp[u * 32];               // coalesced 256B/tile/wave
            const float  w  = a1 * bw[u];
            const float2 yv = yp[u * 4];                // row j+kb+u -> +8 floats
            acc += w * (mv.x * yv.x + mv.y * yv.y);
        }
    }

    atomicAdd(&red[c], acc);                            // ds_add_f32 block reduce
    __syncthreads();
    if (threadIdx.x < 8) atomicAdd(&out[threadIdx.x], red[threadIdx.x]);
}

// ---------------------------------------------------------------------------
// Terms 2+3 via WMMA f32_16x16x32_f16.
// Z[rr,c] = sum_p T[r,c,p]*w[r]*y[jrow,p]; A'[rr,(c',p)] holds the products,
// B is the constant selector B[(c',p), n] = (n == cpass*4 + c').
// 288 units = {term} x {cpass} x {16-row batch}; 144 waves, 2 units each.
// `unit` is an SGPR -> scalar branches only, EXEC stays all-1s for WMMA.
// ---------------------------------------------------------------------------
__global__ void __launch_bounds__(256)
small_terms_kernel(const float* __restrict__ Mt,
                   const float* __restrict__ M,
                   const float* __restrict__ y,
                   const float* __restrict__ wa,
                   const float* __restrict__ wb,
                   float* __restrict__ out)
{
    __shared__ float sY[768];
    for (int x = threadIdx.x; x < 768; x += blockDim.x) sY[x] = y[x];
    __syncthreads();

    const int gid   = blockIdx.x * blockDim.x + threadIdx.x;
    const int gwave = __builtin_amdgcn_readfirstlane(gid >> 5);   // SGPR, 0..143
    const int lane  = threadIdx.x & 31;
    const int rr    = lane & 15;                         // A row / D column
    const int hi    = lane >> 4;                         // lane half -> K block

    v8f acc = {};
    for (int unit = gwave; unit < 288; unit += 144) {    // scalar loop, 2 iters
        const int term  = unit / 144;                    // SGPR: 0=M_tilde/a, 1=M/b
        const int rem   = unit - term * 144;
        const int cpass = rem / 72;                      // SGPR: c 0-3 or 4-7
        const int Rb    = rem - cpass * 72;              // SGPR
        const int r     = Rb * 16 + rr;
        const int c0    = cpass * 4 + hi;                // c for A elems 0-7
        const int c1    = c0 + 2;                        // c for A elems 8-15
        const int jrow  = r % kMd;

        const float* wsrc  = (term == 0) ? wa : wb;               // s_cselect
        const float* Tbase = (term == 0) ? Mt : M;                // s_cselect
        const size_t rstr  = (term == 0) ? (size_t)64 : (size_t)73728;
        const float  w     = wsrc[r];
        const float* T     = Tbase + (size_t)r * rstr;

        const float4 q0 = *reinterpret_cast<const float4*>(T + c0 * 8);
        const float4 q1 = *reinterpret_cast<const float4*>(T + c0 * 8 + 4);
        const float4 q2 = *reinterpret_cast<const float4*>(T + c1 * 8);
        const float4 q3 = *reinterpret_cast<const float4*>(T + c1 * 8 + 4);
        const float4 y0 = *reinterpret_cast<const float4*>(&sY[jrow * 8]);
        const float4 y1 = *reinterpret_cast<const float4*>(&sY[jrow * 8 + 4]);

        v16h A;
        A[0]  = (_Float16)(q0.x * w * y0.x);  A[1]  = (_Float16)(q0.y * w * y0.y);
        A[2]  = (_Float16)(q0.z * w * y0.z);  A[3]  = (_Float16)(q0.w * w * y0.w);
        A[4]  = (_Float16)(q1.x * w * y1.x);  A[5]  = (_Float16)(q1.y * w * y1.y);
        A[6]  = (_Float16)(q1.z * w * y1.z);  A[7]  = (_Float16)(q1.w * w * y1.w);
        A[8]  = (_Float16)(q2.x * w * y0.x);  A[9]  = (_Float16)(q2.y * w * y0.y);
        A[10] = (_Float16)(q2.z * w * y0.z);  A[11] = (_Float16)(q2.w * w * y0.w);
        A[12] = (_Float16)(q3.x * w * y1.x);  A[13] = (_Float16)(q3.y * w * y1.y);
        A[14] = (_Float16)(q3.z * w * y1.z);  A[15] = (_Float16)(q3.w * w * y1.w);

        v16h B;
        #pragma unroll
        for (int e = 0; e < 16; ++e) {
            const int k = (e < 8) ? (hi * 8 + e) : (16 + hi * 8 + (e - 8)); // B row K for this slot
            B[e] = ((lane & 15) == (cpass * 4 + (k >> 3)))
                       ? (_Float16)1.0f : (_Float16)0.0f;
        }

        acc = __builtin_amdgcn_wmma_f32_16x16x32_f16(
                  /*neg_a=*/false, A, /*neg_b=*/false, B,
                  /*c_mod=*/(short)0, acc, /*reuse_a=*/false, /*reuse_b=*/false);
    }

    // D layout: lane holds column n = lane&15; sum its 8 row-slots.
    float s = acc[0] + acc[1] + acc[2] + acc[3] + acc[4] + acc[5] + acc[6] + acc[7];
    if ((lane & 15) < 8) atomicAdd(&out[lane & 15], s);
}

// ---------------------------------------------------------------------------
extern "C" void kernel_launch(void* const* d_in, const int* in_sizes, int n_in,
                              void* d_out, int out_size, void* d_ws, size_t ws_size,
                              hipStream_t stream) {
    const float* y     = (const float*)d_in[0];   // (96, 8, 1)
    const float* Mt    = (const float*)d_in[1];   // (24, 48, 8, 8)
    const float* M     = (const float*)d_in[2];   // (24, 48, 24, 48, 8, 8)
    const float* sigma = (const float*)d_in[3];
    const float* lam   = (const float*)d_in[4];
    const float* phi   = (const float*)d_in[5];   // (48, 24)
    const float* phit  = (const float*)d_in[6];   // (48, 24)
    float* out = (float*)d_out;                   // (8, 1)

    float* wa = (float*)d_ws;                     // 1152 floats
    float* wb = wa + kR;                          // 1152 floats

    // Same stream -> kernels serialize: setup (init out) -> big stream -> finisher.
    setup_kernel<<<5, 256, 0, stream>>>(y, Mt, sigma, lam, phi, phit, wa, wb, out);
    term4_kernel<<<1024, 256, 0, stream>>>(M, y, wa, wb, out);
    small_terms_kernel<<<18, 256, 0, stream>>>(Mt, M, y, wa, wb, out);
}
